// CausalGIN_87823491268858
// MI455X (gfx1250) — matrix operand order repaired and checked by reading, not compiled, not run on hardware
//
#include <hip/hip_runtime.h>
#include <math.h>

#define GDIM 16
#define CDIM 16
#define EPSV 1e-5f

typedef __bf16 bf16_t;
typedef bf16_t v16bf __attribute__((ext_vector_type(16)));
typedef float  v8f  __attribute__((ext_vector_type(8)));

#define ATOMIC_ADD_F32(p, v) __hip_atomic_fetch_add((p), (v), __ATOMIC_RELAXED, __HIP_MEMORY_SCOPE_AGENT)

// ---------------------------------------------------------------------------
// Weight pack: fp32 W[128][128] (K x N row major) -> bf16 WMMA B-fragment
// layout.  Element j of lane L's v16bf holds K = kb*32 + (L>>4)*16 + j,
// N = nt*16 + (L&15).  Packed index = ((nt*4+kb)*32 + lane)*16 + j.
// Optionally scales row K by scl[k] (folded BatchNorm scale).
// ---------------------------------------------------------------------------
__global__ __launch_bounds__(256) void k_pack_b(const float* __restrict__ W,
                                                const float* __restrict__ scl,
                                                bf16_t* __restrict__ dst) {
  int idx  = blockIdx.x * 256 + threadIdx.x;      // 0..16383
  int j    = idx & 15;
  int lane = (idx >> 4) & 31;
  int kb   = (idx >> 9) & 3;
  int nt   = idx >> 11;
  int k    = kb * 32 + (lane >> 4) * 16 + j;
  int n    = nt * 16 + (lane & 15);
  float v  = W[k * 128 + n];
  if (scl) v *= scl[k];
  dst[idx] = (bf16_t)v;
}

// Folded BN shift: out[n] = sum_k sft[k] * W[k][n]  (+ extra[n] if given)
__global__ __launch_bounds__(128) void k_fold_bias(const float* __restrict__ W,
                                                   const float* __restrict__ sft,
                                                   const float* __restrict__ extra,
                                                   float* __restrict__ out) {
  int n = threadIdx.x;
  float a = extra ? extra[n] : 0.0f;
  for (int k = 0; k < 128; ++k) a += sft[k] * W[k * 128 + n];
  out[n] = a;
}

// ---------------------------------------------------------------------------
// WMMA GEMM: out[M x 128] = epilogue(A @ B + bias).  A fp32 (converted to
// bf16 on load), B pre-packed bf16.  One wave computes TM=4 consecutive
// 16x16 M-tiles of one 16-wide N column block, holding the 4 B fragments in
// registers across all tiles (4x B-load reuse, 16 WMMA issues per wave).
// All guards are wave-uniform (EXEC all-ones around WMMA).
// ---------------------------------------------------------------------------
template <int RELU_OUT>
__global__ __launch_bounds__(128) void k_gemm_wmma(
    const float* __restrict__ A, const bf16_t* __restrict__ Bp,
    const float* __restrict__ bias, float* __restrict__ out, int M) {
  const int lane   = threadIdx.x & 31;
  const int wave   = blockIdx.x * 4 + (threadIdx.x >> 5);
  const int nt     = wave & 7;
  const int mt0    = (wave >> 3) * 4;
  if (mt0 * 16 >= M) return;
  const int lane15 = lane & 15;
  const int kg     = lane >> 4;

  v16bf b[4];
#pragma unroll
  for (int kb = 0; kb < 4; ++kb)
    b[kb] = *(const v16bf*)(Bp + ((size_t)((nt * 4 + kb) * 32 + lane) << 4));

  const float bv = bias ? bias[nt * 16 + lane15] : 0.0f;

  for (int t = 0; t < 4; ++t) {
    const int mt = mt0 + t;
    if (mt * 16 >= M) break;                       // wave-uniform
    const float* Ar = A + (size_t)(mt * 16 + lane15) * 128;
    v8f acc;
#pragma unroll
    for (int r = 0; r < 8; ++r) acc[r] = bv;
#pragma unroll
    for (int kb = 0; kb < 4; ++kb) {
      const int kbase = kb * 32 + kg * 16;
      float tv[16];
      *(float4*)(tv + 0)  = *(const float4*)(Ar + kbase);
      *(float4*)(tv + 4)  = *(const float4*)(Ar + kbase + 4);
      *(float4*)(tv + 8)  = *(const float4*)(Ar + kbase + 8);
      *(float4*)(tv + 12) = *(const float4*)(Ar + kbase + 12);
      v16bf a;
#pragma unroll
      for (int j = 0; j < 16; ++j) a[j] = (bf16_t)tv[j];
      acc = __builtin_amdgcn_wmma_f32_16x16x32_bf16(false, a, false, b[kb],
                                                    (short)0, acc, false, false);
    }
    // C/D layout: VGPR r -> M = mt*16 + r + (lane>=16 ? 8:0), N = nt*16+lane15
    float* Or = out + (size_t)(mt * 16 + kg * 8) * 128 + nt * 16 + lane15;
#pragma unroll
    for (int r = 0; r < 8; ++r) {
      float v = acc[r];
      if (RELU_OUT) v = fmaxf(v, 0.0f);
      Or[(size_t)r * 128] = v;
    }
  }
}

// ---------------------------------------------------------------------------
// Per-column (feature) batch statistics: sums / sums-of-squares with atomics.
// ---------------------------------------------------------------------------
__global__ __launch_bounds__(128) void k_colstats(const float* __restrict__ Z,
                                                  float* __restrict__ sums,
                                                  float* __restrict__ sqs, int M) {
  int f  = threadIdx.x;
  int r0 = blockIdx.x * 128;
  int r1 = min(r0 + 128, M);
  float s = 0.f, q = 0.f;
  for (int r = r0; r < r1; ++r) {
    float v = Z[(size_t)r * 128 + f];
    s += v; q += v * v;
  }
  ATOMIC_ADD_F32(&sums[f], s);
  ATOMIC_ADD_F32(&sqs[f], q);
}

__global__ __launch_bounds__(128) void k_bn_finalize(
    const float* __restrict__ sums, const float* __restrict__ sqs,
    const float* __restrict__ g, const float* __restrict__ b, int M,
    float* __restrict__ scl, float* __restrict__ sft) {
  int c = threadIdx.x;
  float mu  = sums[c] / (float)M;
  float var = sqs[c] / (float)M - mu * mu;
  float s   = g[c] * rsqrtf(var + EPSV);
  scl[c] = s;
  sft[c] = b[c] - mu * s;
}

// ---------------------------------------------------------------------------
// Elementwise helpers
// ---------------------------------------------------------------------------
__global__ __launch_bounds__(256) void k_scale_copy(const float* __restrict__ X,
                                                    float coef,
                                                    float* __restrict__ out, int n) {
  int i = blockIdx.x * 256 + threadIdx.x;
  if (i < n) out[i] = coef * X[i];
}

// y = relu(z*scl[f] + sft[f])   (materialized BN+ReLU between conv GEMMs)
__global__ __launch_bounds__(256) void k_affine_relu(const float* __restrict__ Z,
                                                     const float* __restrict__ scl,
                                                     const float* __restrict__ sft,
                                                     float* __restrict__ out, int n) {
  int i = blockIdx.x * 256 + threadIdx.x;
  if (i < n) {
    int f = i & 127;
    out[i] = fmaxf(Z[i] * scl[f] + sft[f], 0.0f);
  }
}

__global__ __launch_bounds__(256) void k_natt_mul(const float* __restrict__ X,
                                                  const float* __restrict__ NA, int s,
                                                  float* __restrict__ out, int n) {
  int i = blockIdx.x * 256 + threadIdx.x;
  if (i < n) out[i] = NA[(i >> 7) * 2 + s] * X[i];
}

__global__ __launch_bounds__(256) void k_gcn_self(const float* __restrict__ Z,
                                                  const float* __restrict__ dis,
                                                  float* __restrict__ H, int n) {
  int i = blockIdx.x * 256 + threadIdx.x;
  if (i < n) {
    float d = dis[i >> 7];
    H[i] = d * d * Z[i];
  }
}

// ---------------------------------------------------------------------------
// Edge scatters (feature-per-lane: one 512B cache line per edge destination)
// ---------------------------------------------------------------------------
__global__ __launch_bounds__(128) void k_gin_scatter(
    const int* __restrict__ row, const int* __restrict__ col, int E,
    const float* __restrict__ X, float coef, float* __restrict__ H) {
  int f  = threadIdx.x;
  int e0 = blockIdx.x * 32;
  int e1 = min(e0 + 32, E);
  for (int e = e0; e < e1; ++e) {
    int r = row[e], c = col[e];
    ATOMIC_ADD_F32(&H[(size_t)c * 128 + f], coef * X[(size_t)r * 128 + f]);
  }
}

__global__ __launch_bounds__(128) void k_gcn_scatter(
    const int* __restrict__ row, const int* __restrict__ col, int E,
    const float* __restrict__ Z, const float* __restrict__ dis,
    const float* __restrict__ EA, int s, float* __restrict__ H) {
  int f  = threadIdx.x;
  int e0 = blockIdx.x * 32;
  int e1 = min(e0 + 32, E);
  for (int e = e0; e < e1; ++e) {
    int r = row[e], c = col[e];
    float w = dis[r] * EA[2 * e + s] * dis[c];
    ATOMIC_ADD_F32(&H[(size_t)c * 128 + f], w * Z[(size_t)r * 128 + f]);
  }
}

__global__ __launch_bounds__(256) void k_deg_init(float* __restrict__ deg, int n) {
  int i = blockIdx.x * 256 + threadIdx.x;
  if (i < n) deg[i] = 1.0f;                         // self loop weight
}

__global__ __launch_bounds__(256) void k_deg_scatter(const int* __restrict__ row,
                                                     const float* __restrict__ EA,
                                                     int s, float* __restrict__ deg,
                                                     int E) {
  int e = blockIdx.x * 256 + threadIdx.x;
  if (e < E) ATOMIC_ADD_F32(&deg[row[e]], EA[2 * e + s]);
}

__global__ __launch_bounds__(256) void k_deg_fin(float* __restrict__ deg, int n) {
  int i = blockIdx.x * 256 + threadIdx.x;
  if (i < n) deg[i] = rsqrtf(deg[i]);               // deg >= 1 always
}

// ---------------------------------------------------------------------------
// Attention heads: wave-per-item dot products + wave32 shuffle reduction
// ---------------------------------------------------------------------------
__global__ __launch_bounds__(256) void k_node_att(const float* __restrict__ X,
                                                  const float* __restrict__ Wn,
                                                  const float* __restrict__ bn2,
                                                  float* __restrict__ NA, int M) {
  int lane = threadIdx.x & 31;
  int wave = blockIdx.x * 8 + (threadIdx.x >> 5);
  if (wave >= M) return;
  const float* xr = X + (size_t)wave * 128;
  float s0 = 0.f, s1 = 0.f;
  for (int j = lane; j < 128; j += 32) {
    float v = xr[j];
    s0 += v * Wn[2 * j];
    s1 += v * Wn[2 * j + 1];
  }
  for (int o = 16; o > 0; o >>= 1) { s0 += __shfl_xor(s0, o); s1 += __shfl_xor(s1, o); }
  if (lane == 0) {
    s0 += bn2[0]; s1 += bn2[1];
    float m  = fmaxf(s0, s1);
    float e0 = __expf(s0 - m), e1 = __expf(s1 - m);
    float inv = 1.0f / (e0 + e1);
    NA[2 * wave] = e0 * inv;
    NA[2 * wave + 1] = e1 * inv;
  }
}

__global__ __launch_bounds__(256) void k_edge_att(const float* __restrict__ X,
                                                  const int* __restrict__ row,
                                                  const int* __restrict__ col,
                                                  const float* __restrict__ We,
                                                  const float* __restrict__ be,
                                                  float* __restrict__ EA, int E) {
  int lane = threadIdx.x & 31;
  int wave = blockIdx.x * 8 + (threadIdx.x >> 5);
  if (wave >= E) return;
  const float* xr = X + (size_t)row[wave] * 128;
  const float* xc = X + (size_t)col[wave] * 128;
  float s0 = 0.f, s1 = 0.f;
  for (int j = lane; j < 128; j += 32) {
    float vr = xr[j], vc = xc[j];
    s0 += vr * We[2 * j]     + vc * We[2 * (128 + j)];
    s1 += vr * We[2 * j + 1] + vc * We[2 * (128 + j) + 1];
  }
  for (int o = 16; o > 0; o >>= 1) { s0 += __shfl_xor(s0, o); s1 += __shfl_xor(s1, o); }
  if (lane == 0) {
    s0 += be[0]; s1 += be[1];
    float m  = fmaxf(s0, s1);
    float e0 = __expf(s0 - m), e1 = __expf(s1 - m);
    float inv = 1.0f / (e0 + e1);
    EA[2 * wave] = e0 * inv;
    EA[2 * wave + 1] = e1 * inv;
  }
}

// ---------------------------------------------------------------------------
// Pool: out[g][f] += relu(V[r][f] + bias[f]) over batch[r]==g.  LDS-private
// 16x128 accumulator per block (column-private per thread -> no LDS atomics).
// ---------------------------------------------------------------------------
__global__ __launch_bounds__(128) void k_pool(const float* __restrict__ V,
                                              const float* __restrict__ bias,
                                              const int* __restrict__ batch,
                                              float* __restrict__ pool, int M) {
  __shared__ float acc[GDIM * 128];
  int f = threadIdx.x;
  float bv = bias[f];
  for (int g = 0; g < GDIM; ++g) acc[g * 128 + f] = 0.f;
  int r0 = blockIdx.x * 256, r1 = min(r0 + 256, M);
  for (int r = r0; r < r1; ++r) {
    int g = batch[r];
    acc[g * 128 + f] += fmaxf(V[(size_t)r * 128 + f] + bv, 0.f);
  }
  for (int g = 0; g < GDIM; ++g)
    ATOMIC_ADD_F32(&pool[g * 128 + f], acc[g * 128 + f]);
}

__global__ __launch_bounds__(256) void k_concat(const float* __restrict__ PC,
                                                const float* __restrict__ PO,
                                                const int* __restrict__ perm,
                                                float* __restrict__ cat) {
  int idx = blockIdx.x * 256 + threadIdx.x;
  if (idx >= GDIM * 256) return;
  int g = idx >> 8, c = idx & 255;
  cat[idx] = (c < 128) ? PC[perm[g] * 128 + c] : PO[g * 128 + (c - 128)];
}

// ---------------------------------------------------------------------------
// Readout head: bn -> @W1+bv1 -> relu -> bn -> @W2+bv2 -> log_softmax.
// G=16 rows, single workgroup, all intermediates in LDS.
// ---------------------------------------------------------------------------
__global__ __launch_bounds__(256) void k_readout(
    const float* __restrict__ Zin, int din,
    const float* __restrict__ g1, const float* __restrict__ bb1,
    const float* __restrict__ W1, const float* __restrict__ bv1,
    const float* __restrict__ g2, const float* __restrict__ bb2,
    const float* __restrict__ W2, const float* __restrict__ bv2,
    float* __restrict__ out) {
  __shared__ float s1[GDIM * 256];
  __shared__ float s2[GDIM * 128];
  int tid = threadIdx.x;
  for (int c = tid; c < din; c += 256) {
    float fr[GDIM];
    float s = 0.f, q = 0.f;
    for (int r = 0; r < GDIM; ++r) {
      float v = Zin[r * din + c];
      fr[r] = v; s += v; q += v * v;
    }
    float mu  = s / (float)GDIM;
    float var = q / (float)GDIM - mu * mu;
    float sc  = g1[c] * rsqrtf(var + EPSV);
    float sh  = bb1[c] - mu * sc;
    for (int r = 0; r < GDIM; ++r) s1[r * din + c] = fr[r] * sc + sh;
  }
  __syncthreads();
  for (int idx = tid; idx < GDIM * 128; idx += 256) {
    int r = idx >> 7, h = idx & 127;
    float a = bv1[h];
    for (int k = 0; k < din; ++k) a += s1[r * din + k] * W1[k * 128 + h];
    s2[idx] = fmaxf(a, 0.f);
  }
  __syncthreads();
  for (int c = tid; c < 128; c += 256) {
    float s = 0.f, q = 0.f;
    for (int r = 0; r < GDIM; ++r) { float v = s2[r * 128 + c]; s += v; q += v * v; }
    float mu  = s / (float)GDIM;
    float var = q / (float)GDIM - mu * mu;
    float sc  = g2[c] * rsqrtf(var + EPSV);
    float sh  = bb2[c] - mu * sc;
    for (int r = 0; r < GDIM; ++r) s2[r * 128 + c] = s2[r * 128 + c] * sc + sh;
  }
  __syncthreads();
  for (int idx = tid; idx < GDIM * CDIM; idx += 256) {
    int r = idx >> 4, c = idx & 15;
    float a = bv2[c];
    for (int k = 0; k < 128; ++k) a += s2[r * 128 + k] * W2[k * CDIM + c];
    s1[idx] = a;
  }
  __syncthreads();
  if (tid < GDIM) {
    float m = -1e30f;
    for (int c = 0; c < CDIM; ++c) m = fmaxf(m, s1[tid * CDIM + c]);
    float s = 0.f;
    for (int c = 0; c < CDIM; ++c) s += __expf(s1[tid * CDIM + c] - m);
    float l = m + __logf(s);
    for (int c = 0; c < CDIM; ++c) out[tid * CDIM + c] = s1[tid * CDIM + c] - l;
  }
}

// ---------------------------------------------------------------------------
// Host orchestration
// ---------------------------------------------------------------------------
extern "C" void kernel_launch(void* const* d_in, const int* in_sizes, int n_in,
                              void* d_out, int out_size, void* d_ws, size_t ws_size,
                              hipStream_t stream) {
  const float* x    = (const float*)d_in[0];
  const int* ei     = (const int*)d_in[1];
  const int* batch  = (const int*)d_in[2];
  const int* perm   = (const int*)d_in[3];
  const int E       = in_sizes[1] / 2;
  const int* row    = ei;
  const int* col    = ei + E;
  const int M       = in_sizes[0] / 128;            // = 50000
  float* out        = (float*)d_out;

  auto P = [&](int i) { return (const float*)d_in[i]; };

  // --- workspace carve (256B aligned) ---
  char* w = (char*)d_ws;
  auto take = [&](size_t bytes) {
    void* p = (void*)w;
    w += (bytes + 255) & ~(size_t)255;
    return p;
  };
  const size_t NF = (size_t)M * 128;
  float*  X    = (float*)take(NF * 4);
  float*  H    = (float*)take(NF * 4);
  float*  Z    = (float*)take(NF * 4);
  float*  EA   = (float*)take((size_t)E * 2 * 4);
  float*  NA   = (float*)take((size_t)M * 2 * 4);
  float*  DEG  = (float*)take((size_t)M * 4);
  float*  SUMS = (float*)take(512);
  float*  SQS  = (float*)take(512);
  float*  SC   = (float*)take(512);
  float*  SH   = (float*)take(512);
  float*  BF   = (float*)take(512);                 // folded bias vector
  float*  PC   = (float*)take(GDIM * 128 * 4);
  float*  PO   = (float*)take(GDIM * 128 * 4);
  float*  CAT  = (float*)take(GDIM * 256 * 4);
  bf16_t* PKs  = (bf16_t*)take((size_t)6 * 16384 * 2);  // static: conv W1/W2 x3
  bf16_t* PKd  = (bf16_t*)take((size_t)16384 * 2);      // dynamic scaled pack

  // --- static packs: conv W1/W2 (no BN fold needed on those) ---
  // slots: 0/1=conv0 W1/W2, 2/3=conv1, 4/5=conv2
  const int widx[6] = {7, 11, 13, 17, 19, 23};
  for (int i = 0; i < 6; ++i)
    k_pack_b<<<64, 256, 0, stream>>>(P(widx[i]), nullptr, PKs + (size_t)i * 16384);

  const int mtiles     = (M + 15) / 16;
  const int gemmBlocks = ((((mtiles + 3) / 4) * 8) + 3) / 4;  // TM=4, 4 waves/blk
  const int ewBlocks   = (int)((NF + 255) / 256);
  const int statBlocks = (M + 127) / 128;

  auto colstats = [&](const float* Zp, const float* g, const float* b) {
    hipMemsetAsync(SUMS, 0, 512, stream);
    hipMemsetAsync(SQS, 0, 512, stream);
    k_colstats<<<statBlocks, 128, 0, stream>>>(Zp, SUMS, SQS, M);
    k_bn_finalize<<<1, 128, 0, stream>>>(SUMS, SQS, g, b, M, SC, SH);
  };
  // pack diag(SC)@W into PKd and BF = SH@W (+extra): bn fully folded into GEMM
  auto fold_pack = [&](const float* W, const float* extra) {
    k_pack_b<<<64, 256, 0, stream>>>(W, SC, PKd);
    k_fold_bias<<<1, 128, 0, stream>>>(W, SH, extra, BF);
  };

  // --- feature BN folded into first GEMM: X = relu(bn(x) @ Wf) ---
  colstats(x, P(4), P(5));
  fold_pack(P(6), nullptr);
  k_gemm_wmma<1><<<gemmBlocks, 128, 0, stream>>>(x, PKd, BF, X, M);

  // --- GIN conv layers ---
  for (int i = 0; i < 3; ++i) {
    const int base = 7 + i * 6;  // W1,b1,g,bbn,W2,b2
    const float coef = (i == 0) ? 1.0f : 2.0f;
    k_scale_copy<<<ewBlocks, 256, 0, stream>>>(X, coef, H, (int)NF);
    k_gin_scatter<<<(E + 31) / 32, 128, 0, stream>>>(row, col, E, X, coef, H);
    // z = h @ W1   (b1 omitted: column-constant bias cancels exactly in BN)
    k_gemm_wmma<0><<<gemmBlocks, 128, 0, stream>>>(
        H, PKs + (size_t)(2 * i) * 16384, nullptr, Z, M);
    colstats(Z, P(base + 2), P(base + 3));
    k_affine_relu<<<ewBlocks, 256, 0, stream>>>(Z, SC, SH, H, (int)NF);
    // x = relu(h' @ W2 + b2)
    k_gemm_wmma<1><<<gemmBlocks, 128, 0, stream>>>(
        H, PKs + (size_t)(2 * i + 1) * 16384, P(base + 5), X, M);
  }

  // --- attention heads ---
  k_node_att<<<(M + 7) / 8, 256, 0, stream>>>(X, P(27), P(28), NA, M);
  k_edge_att<<<(E + 7) / 8, 256, 0, stream>>>(X, row, col, P(25), P(26), EA, E);

  // --- two GCN branches (s=0: causal, s=1: other) ---
  for (int s = 0; s < 2; ++s) {
    const float* bng  = (s == 0) ? P(29) : P(31);
    const float* bnb  = (s == 0) ? P(30) : P(32);
    const float* W    = (s == 0) ? P(33) : P(35);
    const float* bias = (s == 0) ? P(34) : P(36);
    float*      POOL  = (s == 0) ? PC : PO;

    k_natt_mul<<<ewBlocks, 256, 0, stream>>>(X, NA, s, H, (int)NF);
    colstats(H, bng, bnb);
    fold_pack(W, nullptr);
    // z = bn(x_s) @ W  (BN fully folded into packed B + bias vector)
    k_gemm_wmma<0><<<gemmBlocks, 128, 0, stream>>>(H, PKd, BF, Z, M);
    // symmetric degree norm with softmax edge weights + self loops
    k_deg_init<<<(M + 255) / 256, 256, 0, stream>>>(DEG, M);
    k_deg_scatter<<<(E + 255) / 256, 256, 0, stream>>>(row, EA, s, DEG, E);
    k_deg_fin<<<(M + 255) / 256, 256, 0, stream>>>(DEG, M);
    // out = dis^2 * z (self loop) + scatter(dis_r * ew * dis_c * z_r -> c)
    k_gcn_self<<<ewBlocks, 256, 0, stream>>>(Z, DEG, H, (int)NF);
    k_gcn_scatter<<<(E + 31) / 32, 128, 0, stream>>>(row, col, E, Z, DEG, EA, s, H);
    // pool with fused (+bias, relu)
    hipMemsetAsync(POOL, 0, GDIM * 128 * 4, stream);
    k_pool<<<(M + 255) / 256, 128, 0, stream>>>(H, bias, batch, POOL, M);
  }

  // --- readouts: (xc_logis, xo_logis, xco_logis) -> d_out[0:256|256:512|512:768]
  k_readout<<<1, 256, 0, stream>>>(PC, 128, P(37), P(38), P(39), P(40),
                                   P(41), P(42), P(43), P(44), out + 0);
  k_readout<<<1, 256, 0, stream>>>(PO, 128, P(45), P(46), P(47), P(48),
                                   P(49), P(50), P(51), P(52), out + 256);
  k_concat<<<16, 256, 0, stream>>>(PC, PO, perm, CAT);
  k_readout<<<1, 256, 0, stream>>>(CAT, 256, P(53), P(54), P(55), P(56),
                                   P(57), P(58), P(59), P(60), out + 512);
}
